// RWKV_TMix_x060_17368847745422
// MI455X (gfx1250) — matrix operand order repaired and checked by reading, not compiled, not run on hardware
//
#include <hip/hip_runtime.h>
#include <hip/hip_bf16.h>

// ---------------------------------------------------------------------------
// RWKV6 time-mix for gfx1250 (MI455X). bf16 WMMA GEMMs + fp32 WKV scan.
// GEMM: each wave computes a 32x64 output block (2 M-tiles x 4 N-tiles),
// reusing B fragments across the two M-tiles -> 8 wmma per 12 b128 loads.
// ---------------------------------------------------------------------------

typedef __attribute__((ext_vector_type(16))) __bf16 v16bf;
typedef __attribute__((ext_vector_type(8)))  float  v8f;
typedef __attribute__((ext_vector_type(4)))  unsigned int u32x4;

constexpr int Bsz  = 4;
constexpr int Tlen = 2048;
constexpr int Cdim = 1024;
constexpr int Hh   = 16;
constexpr int Kh   = 64;
constexpr int NT   = Bsz * Tlen;          // 8192 tokens
constexpr int EMIX = 32;
constexpr int EDEC = 64;
constexpr float GN_EPS = 1e-5f * 64.0f;   // eps * head_size_divisor^2

__device__ __forceinline__ unsigned short f2bf(float f) {
    unsigned int u = __float_as_uint(f);
    unsigned int r = u + 0x7FFFu + ((u >> 16) & 1u);   // round-to-nearest-even
    return (unsigned short)(r >> 16);
}

// ------------------------------- fp32 -> bf16 ------------------------------
__global__ __launch_bounds__(256) void cvt_bf16_kernel(const float* __restrict__ in,
                                                       unsigned short* __restrict__ out,
                                                       int n) {
    int i = blockIdx.x * 256 + threadIdx.x;
    if (i < n) out[i] = f2bf(in[i]);
}

// ------------------------------- prep kernel -------------------------------
// One block (256 threads) per token: token-shift, low-rank data-dependent
// mixing, emit xr/xk/xv/xg (bf16) and decay (fp32).
__global__ __launch_bounds__(256) void prep_kernel(
    const float* __restrict__ x,
    const float* __restrict__ tmx, const float* __restrict__ tmw,
    const float* __restrict__ tmk, const float* __restrict__ tmv,
    const float* __restrict__ tmr, const float* __restrict__ tmg,
    const float* __restrict__ maa_w1,   // [C,160]
    const float* __restrict__ maa_w2,   // [5,32,C]
    const float* __restrict__ tdec,     // [C]
    const float* __restrict__ td_w1,    // [C,64]
    const float* __restrict__ td_w2,    // [64,C]
    unsigned short* __restrict__ xr_o, unsigned short* __restrict__ xk_o,
    unsigned short* __restrict__ xv_o, unsigned short* __restrict__ xg_o,
    float* __restrict__ decay_o) {

    __shared__ float sx[Cdim], sxx[Cdim], sxm[Cdim], sxw[Cdim];
    __shared__ float sh1[5 * EMIX];
    __shared__ float sh2[EDEC];

    const int bt  = blockIdx.x;            // 0..NT-1
    const int t   = bt & (Tlen - 1);
    const int tid = threadIdx.x;
    const size_t rowOff = (size_t)bt * Cdim;

    // token shift, and maa_x-mixed input
    for (int c = tid; c < Cdim; c += 256) {
        float xc = x[rowOff + c];
        float xp = (t == 0) ? 0.0f : x[rowOff - Cdim + c];
        float d  = xp - xc;                 // xx
        sx[c]  = xc;
        sxx[c] = d;
        sxm[c] = xc + d * tmx[c];
    }
    __syncthreads();

    // h1 = tanh(xm @ maa_w1) : 160 columns
    for (int e = tid; e < 5 * EMIX; e += 256) {
        float s = 0.0f;
        for (int c = 0; c < Cdim; ++c) s += sxm[c] * maa_w1[(size_t)c * 160 + e];
        sh1[e] = tanhf(s);
    }
    __syncthreads();

    // per-channel low-rank corrections and the five mixed streams
    for (int c = tid; c < Cdim; c += 256) {
        float m0 = 0, m1 = 0, m2 = 0, m3 = 0, m4 = 0;
        for (int e = 0; e < EMIX; ++e) {
            m0 += sh1[0 * EMIX + e] * maa_w2[(size_t)(0 * EMIX + e) * Cdim + c];
            m1 += sh1[1 * EMIX + e] * maa_w2[(size_t)(1 * EMIX + e) * Cdim + c];
            m2 += sh1[2 * EMIX + e] * maa_w2[(size_t)(2 * EMIX + e) * Cdim + c];
            m3 += sh1[3 * EMIX + e] * maa_w2[(size_t)(3 * EMIX + e) * Cdim + c];
            m4 += sh1[4 * EMIX + e] * maa_w2[(size_t)(4 * EMIX + e) * Cdim + c];
        }
        float xc = sx[c], d = sxx[c];
        sxw[c] = xc + d * (tmw[c] + m0);
        xk_o[rowOff + c] = f2bf(xc + d * (tmk[c] + m1));
        xv_o[rowOff + c] = f2bf(xc + d * (tmv[c] + m2));
        xr_o[rowOff + c] = f2bf(xc + d * (tmr[c] + m3));
        xg_o[rowOff + c] = f2bf(xc + d * (tmg[c] + m4));
    }
    __syncthreads();

    // h2 = tanh(xw @ td_w1)
    for (int e = tid; e < EDEC; e += 256) {
        float s = 0.0f;
        for (int c = 0; c < Cdim; ++c) s += sxw[c] * td_w1[(size_t)c * EDEC + e];
        sh2[e] = tanhf(s);
    }
    __syncthreads();

    // decay = exp(-exp(time_decay + h2 @ td_w2))
    for (int c = tid; c < Cdim; c += 256) {
        float s = tdec[c];
        for (int e = 0; e < EDEC; ++e) s += sh2[e] * td_w2[(size_t)e * Cdim + c];
        decay_o[rowOff + c] = expf(-expf(s));
    }
}

// ------------------------------- WMMA GEMM ---------------------------------
// Out[m,n] = sum_k A[m,k] * W[n,k]   (A: [NT x 1024] bf16, W: [1024 x 1024] bf16)
// Block = 256 threads (8 waves). Wave -> 32x64 output block (2 M-tiles x
// 4 N-tiles; B fragments reused across both M-tiles).
// grid = (NT/64, Cdim/256).
union FragA { v16bf v; u32x4 q[2]; };

__global__ __launch_bounds__(256) void gemm_bf16_nt(
    const unsigned short* __restrict__ A,
    const unsigned short* __restrict__ W,
    float* __restrict__ Out) {

    constexpr int Kd = Cdim, Nd = Cdim;
    const int tid  = threadIdx.x;
    const int lane = tid & 31;
    const int wave = tid >> 5;
    const int wm   = wave & 1;          // 0..1  (M half of the block)
    const int wn   = wave >> 1;         // 0..3  (N group)

    const int mBase = blockIdx.x * 64 + wm * 32;        // two 16-row tiles
    const int nBase = blockIdx.y * 256 + wn * 64;       // first output column

    const int r16 = lane & 15;          // M row within tile / N row within tile
    const int hi  = lane >> 4;          // half-wave select
    const int aShift = hi * 16;         // A: contiguous 16 K values
    const int bShift = hi * 16;         // B: contiguous 16 K values

    const unsigned short* aRow0 = A + (size_t)(mBase + r16) * Kd + aShift;
    const unsigned short* aRow1 = aRow0 + (size_t)16 * Kd;

    v8f acc0[4] = {};
    v8f acc1[4] = {};
    for (int kb = 0; kb < Kd; kb += 32) {
        FragA a0, a1;
        a0.q[0] = *(const u32x4*)(aRow0 + kb);
        a0.q[1] = *(const u32x4*)(aRow0 + kb + 8);
        a1.q[0] = *(const u32x4*)(aRow1 + kb);
        a1.q[1] = *(const u32x4*)(aRow1 + kb + 8);
#pragma unroll
        for (int i = 0; i < 4; ++i) {
            const unsigned short* bRow =
                W + (size_t)(nBase + i * 16 + r16) * Kd + kb + bShift;
            v16bf bv = *(const v16bf*)bRow;
            acc0[i] = __builtin_amdgcn_wmma_f32_16x16x32_bf16(
                false, a0.v, false, bv, (short)0, acc0[i], false, false);
            acc1[i] = __builtin_amdgcn_wmma_f32_16x16x32_bf16(
                false, a1.v, false, bv, (short)0, acc1[i], false, false);
        }
    }

    // C/D layout: VGPR j -> M = j + 8*hi ; column N = lane&15
    const int mOut0 = mBase + hi * 8;
    const int nCol  = lane & 15;
#pragma unroll
    for (int i = 0; i < 4; ++i) {
#pragma unroll
        for (int j = 0; j < 8; ++j) {
            Out[(size_t)(mOut0 + j) * Nd + nBase + i * 16 + nCol]      = acc0[i][j];
            Out[(size_t)(mOut0 + 16 + j) * Nd + nBase + i * 16 + nCol] = acc1[i][j];
        }
    }
}

// ------------------------------- WKV6 scan ---------------------------------
// One block per (b,h); 64 threads; thread j owns state column S[k][j].
// Fused GroupNorm (per head) + silu(g) + ln affine; emits (y*g) as bf16.
__global__ __launch_bounds__(64) void wkv_kernel(
    const float* __restrict__ r, const float* __restrict__ k,
    const float* __restrict__ v, const float* __restrict__ decay,
    const float* __restrict__ graw, const float* __restrict__ u,
    const float* __restrict__ ln_g, const float* __restrict__ ln_b,
    unsigned short* __restrict__ yg) {

    const int bh = blockIdx.x;
    const int b  = bh >> 4;             // H = 16
    const int h  = bh & 15;
    const int j  = threadIdx.x;         // value column 0..63

    __shared__ float rL[Kh], kL[Kh], dL[Kh], uL[Kh], yL[Kh];

    const int ch = h * Kh + j;
    uL[j] = u[ch];
    const float lng = ln_g[ch];
    const float lnb = ln_b[ch];

    float S[Kh];
#pragma unroll
    for (int kk = 0; kk < Kh; ++kk) S[kk] = 0.0f;
    __syncthreads();

    for (int t = 0; t < Tlen; ++t) {
        const size_t idx = ((size_t)b * Tlen + t) * Cdim + ch;
        rL[j] = r[idx];
        kL[j] = k[idx];
        dL[j] = decay[idx];
        const float vj = v[idx];
        __syncthreads();

        float y = 0.0f;
#pragma unroll
        for (int kk = 0; kk < Kh; ++kk) {
            float kv = kL[kk] * vj;
            y += rL[kk] * (S[kk] + uL[kk] * kv);
            S[kk] = dL[kk] * S[kk] + kv;
        }
        yL[j] = y;
        __syncthreads();

        float mu = 0.0f;
#pragma unroll
        for (int kk = 0; kk < Kh; ++kk) mu += yL[kk];
        mu *= (1.0f / Kh);
        float var = 0.0f;
#pragma unroll
        for (int kk = 0; kk < Kh; ++kk) {
            float d = yL[kk] - mu;
            var += d * d;
        }
        var *= (1.0f / Kh);

        float yn = (y - mu) * rsqrtf(var + GN_EPS) * lng + lnb;
        float gv = graw[idx];
        float g  = gv / (1.0f + expf(-gv));   // silu
        yg[idx]  = f2bf(yn * g);
        __syncthreads();
    }
}

// ------------------------------- launcher ----------------------------------
extern "C" void kernel_launch(void* const* d_in, const int* in_sizes, int n_in,
                              void* d_out, int out_size, void* d_ws, size_t ws_size,
                              hipStream_t stream) {
    const float* x      = (const float*)d_in[0];
    const float* tmx    = (const float*)d_in[1];
    const float* tmw    = (const float*)d_in[2];
    const float* tmk    = (const float*)d_in[3];
    const float* tmv    = (const float*)d_in[4];
    const float* tmr    = (const float*)d_in[5];
    const float* tmg    = (const float*)d_in[6];
    const float* maa_w1 = (const float*)d_in[7];
    const float* maa_w2 = (const float*)d_in[8];
    const float* tdec   = (const float*)d_in[9];
    const float* td_w1  = (const float*)d_in[10];
    const float* td_w2  = (const float*)d_in[11];
    const float* u      = (const float*)d_in[12];
    const float* Wr     = (const float*)d_in[13];
    const float* Wk     = (const float*)d_in[14];
    const float* Wv     = (const float*)d_in[15];
    const float* Wg     = (const float*)d_in[16];
    const float* Wo     = (const float*)d_in[17];
    const float* ln_g   = (const float*)d_in[18];
    const float* ln_b   = (const float*)d_in[19];
    float* out = (float*)d_out;

    // ---- workspace layout ----
    char* wsb = (char*)d_ws;
    size_t off = 0;
    auto alloc = [&](size_t bytes) { void* p = wsb + off; off += (bytes + 255) & ~(size_t)255; return p; };

    const size_t tokElems = (size_t)NT * Cdim;   // 8M
    const size_t wElems   = (size_t)Cdim * Cdim; // 1M

    unsigned short* xrB = (unsigned short*)alloc(tokElems * 2);
    unsigned short* xkB = (unsigned short*)alloc(tokElems * 2);
    unsigned short* xvB = (unsigned short*)alloc(tokElems * 2);
    unsigned short* xgB = (unsigned short*)alloc(tokElems * 2);
    unsigned short* WrB = (unsigned short*)alloc(wElems * 2);
    unsigned short* WkB = (unsigned short*)alloc(wElems * 2);
    unsigned short* WvB = (unsigned short*)alloc(wElems * 2);
    unsigned short* WgB = (unsigned short*)alloc(wElems * 2);
    unsigned short* WoB = (unsigned short*)alloc(wElems * 2);
    float* rF    = (float*)alloc(tokElems * 4);
    float* kF    = (float*)alloc(tokElems * 4);
    float* vF    = (float*)alloc(tokElems * 4);
    float* gF    = (float*)alloc(tokElems * 4);
    float* decF  = (float*)alloc(tokElems * 4);
    unsigned short* ygB = (unsigned short*)alloc(tokElems * 2);
    (void)ws_size; (void)in_sizes; (void)n_in; (void)out_size;

    // 1) weights fp32 -> bf16
    {
        int n = (int)wElems;
        int blocks = (n + 255) / 256;
        cvt_bf16_kernel<<<blocks, 256, 0, stream>>>(Wr, WrB, n);
        cvt_bf16_kernel<<<blocks, 256, 0, stream>>>(Wk, WkB, n);
        cvt_bf16_kernel<<<blocks, 256, 0, stream>>>(Wv, WvB, n);
        cvt_bf16_kernel<<<blocks, 256, 0, stream>>>(Wg, WgB, n);
        cvt_bf16_kernel<<<blocks, 256, 0, stream>>>(Wo, WoB, n);
    }

    // 2) token-shift + low-rank mixing + decay
    prep_kernel<<<NT, 256, 0, stream>>>(x, tmx, tmw, tmk, tmv, tmr, tmg,
                                        maa_w1, maa_w2, tdec, td_w1, td_w2,
                                        xrB, xkB, xvB, xgB, decF);

    // 3) four WMMA GEMMs: r, k, v, g(raw)
    dim3 ggrid(NT / 64, Cdim / 256);
    gemm_bf16_nt<<<ggrid, 256, 0, stream>>>(xrB, WrB, rF);
    gemm_bf16_nt<<<ggrid, 256, 0, stream>>>(xkB, WkB, kF);
    gemm_bf16_nt<<<ggrid, 256, 0, stream>>>(xvB, WvB, vF);
    gemm_bf16_nt<<<ggrid, 256, 0, stream>>>(xgB, WgB, gF);

    // 4) WKV6 recurrence + GroupNorm + silu(g), emits bf16 (y*g)
    wkv_kernel<<<Bsz * Hh, Kh, 0, stream>>>(rF, kF, vF, decF, gF, u, ln_g, ln_b, ygB);

    // 5) output projection (y*g) @ Wo^T -> fp32 d_out
    gemm_bf16_nt<<<ggrid, 256, 0, stream>>>(ygB, WoB, out);
}
